// LinearTransformer_24936580121040
// MI455X (gfx1250) — matrix operand order repaired and checked by reading, not compiled, not run on hardware
//
#include <hip/hip_runtime.h>

typedef __attribute__((ext_vector_type(16))) _Float16 v16h;
typedef __attribute__((ext_vector_type(8)))  _Float16 v8h;
typedef __attribute__((ext_vector_type(8)))  float    v8f;

#define NBATCH 4
#define NSEQ   65536
#define NWAVE  4              // waves per block (128 threads)
#define BLKX   128            // blocks per batch
#define WPB    (BLKX*NWAVE)   // 512 waves per batch
#define NTILE  (NSEQ/32)      // 2048 32-token tiles per batch
#define PART_F 288            // 256 S + 32 Z partial floats per wave

// D = A(16x32 f16) * B(32x16 f16) + C(f32)
__device__ __forceinline__ v8f wmma_f16(v16h a, v16h b, v8f c) {
  return __builtin_amdgcn_wmma_f32_16x16x32_f16(false, a, false, b, (short)0, c, false, false);
}

// A-fragment K mapping: lane<16: halves j<8 -> K=j, j>=8 -> K=j+8 ; lane>=16: +8
__device__ __forceinline__ int jkA(int j, int l) { return (j < 8 ? j : j + 8) + ((l & 16) ? 8 : 0); }

// Gather A fragment from LDS stage: row-major [rows][stride] halves, contiguous 8-half chunks
__device__ __forceinline__ v16h load_a_frag(const _Float16* s, int row, int stride, int k0, int l) {
  const _Float16* p = s + row * stride + k0 + ((l & 16) ? 8 : 0);
  v8h lo = *(const v8h*)p;
  v8h hi = *(const v8h*)(p + 16);
  return __builtin_shufflevector(lo, hi, 0,1,2,3,4,5,6,7,8,9,10,11,12,13,14,15);
}

__device__ __forceinline__ void layer_norm32(const float* x, float* y, const float* g, const float* b) {
  float m = 0.f;
#pragma unroll
  for (int d = 0; d < 32; ++d) m += x[d];
  m *= (1.f / 32.f);
  float v = 0.f;
#pragma unroll
  for (int d = 0; d < 32; ++d) { float t = x[d] - m; v += t * t; }
  v *= (1.f / 32.f);
  float rs = rsqrtf(v + 1e-5f);
#pragma unroll
  for (int d = 0; d < 32; ++d) y[d] = (x[d] - m) * rs * g[d] + b[d];
}

__device__ __forceinline__ void store_f16_row(_Float16* dst, const float* y) {
#pragma unroll
  for (int c = 0; c < 4; ++c) {
    v8h p;
#pragma unroll
    for (int i = 0; i < 8; ++i) p[i] = (_Float16)y[c * 8 + i];
    *(v8h*)(dst + c * 8) = p;
  }
}

// ---------------- embed: h[t][d] = x[t]*w_in[d] + b_in[d] ----------------
__global__ __launch_bounds__(256) void k_embed(const float* __restrict__ x,
                                               const float* __restrict__ win,
                                               const float* __restrict__ bin,
                                               float* __restrict__ hbuf) {
  size_t i = (size_t)blockIdx.x * 256 + threadIdx.x;
  if (i < (size_t)NBATCH * NSEQ * 32) {
    int d = (int)(i & 31);
    hbuf[i] = x[i >> 5] * win[d] + bin[d];
  }
}

// ---------------- per-layer: partial ctx reduction ----------------
__global__ __launch_bounds__(128) void k_ctx(const float* __restrict__ hbuf,
    const float* __restrict__ g1p, const float* __restrict__ b1p,
    const float* __restrict__ wk, const float* __restrict__ wv,
    float* __restrict__ part) {
  __shared__ v16h wfr[2][32];
  __shared__ __align__(16) _Float16 ystage[NWAVE][1024];
  __shared__ __align__(16) _Float16 kvstage[NWAVE][1024];
  const int tid = threadIdx.x;
  const int b = blockIdx.y;
  // B fragments for wk, wv (32x16 row-major): lane holds col n=l&15, half j -> K = (j<8 ? j : j+8) + 8*(l>>4)
  for (int i = tid; i < 2 * 512; i += 128) {
    int f = i >> 9, e = i & 511, l2 = e >> 4, j = e & 15;
    int kk = (j < 8 ? j : j + 8) + ((l2 & 16) ? 8 : 0);
    const float* W = f ? wv : wk;
    ((_Float16*)&wfr[f][l2])[j] = (_Float16)W[kk * 16 + (l2 & 15)];
  }
  __syncthreads();
  const int w = tid >> 5, l = tid & 31, col = l & 15, rb = (l >> 4) << 3;
  _Float16* ys = ystage[w];
  _Float16* kvs = kvstage[w];
  float g1[32], bb1[32];
#pragma unroll
  for (int d = 0; d < 32; ++d) { g1[d] = g1p[d]; bb1[d] = b1p[d]; }
  const v16h bk = wfr[0][l], bv = wfr[1][l];
  v8f ctxS = {};
  float zp = 0.f;
  const int wg = blockIdx.x * NWAVE + w;
  for (int t = wg; t < NTILE; t += WPB) {       // exactly 4 uniform iterations
    const float* hp = hbuf + ((size_t)b * NSEQ + (size_t)t * 32 + l) * 32;
    float h0[32], y[32];
#pragma unroll
    for (int d = 0; d < 32; ++d) h0[d] = hp[d];
    layer_norm32(h0, y, g1, bb1);
    store_f16_row(ys + l * 32, y);
    __syncthreads();
#pragma unroll
    for (int mt = 0; mt < 2; ++mt) {
      v16h a = load_a_frag(ys, mt * 16 + col, 32, 0, l);
      v8f zc = {};
      v8f ka = wmma_f16(a, bk, zc);   // k = y @ wk
      v8f va = wmma_f16(a, bv, zc);   // v = y @ wv
#pragma unroll
      for (int r = 0; r < 8; ++r) {   // D layout: row m = rb+r, col = l&15
        float ek = __expf(ka[r]);     // native v_exp_f32 (args are O(0.1))
        zp += ek;                     // Z partial for feature d=col
        kvs[(mt * 16 + rb + r) * 32 + col]      = (_Float16)ek;
        kvs[(mt * 16 + rb + r) * 32 + 16 + col] = (_Float16)va[r];
      }
    }
    __syncthreads();
    // S += ek^T (16d x 32t) @ v (32t x 16e), accumulated in WMMA C across tiles
    v16h ae, bvf;
#pragma unroll
    for (int j = 0; j < 16; ++j) {
      ae[j]  = kvs[jkA(j, l) * 32 + col];                       // A[d=col][t]
      bvf[j] = kvs[jkA(j, l) * 32 + 16 + col];  // B[t][e=col]
    }
    ctxS = wmma_f16(ae, bvf, ctxS);
  }
  float* pp = part + ((size_t)b * WPB + wg) * PART_F;
#pragma unroll
  for (int r = 0; r < 8; ++r) pp[(rb + r) * 16 + col] = ctxS[r];  // S[d=rb+r][e=col]
  pp[256 + l] = zp;
}

// ---------------- per-layer: finalize ctx = S/Z, emit as B-fragment ----------------
__global__ __launch_bounds__(256) void k_fin(const float* __restrict__ part,
                                             _Float16* __restrict__ ctxf) {
  const int b = blockIdx.x, t = threadIdx.x;
  __shared__ float ctx[16][16];
  const int d = t >> 4, e = t & 15;
  float S = 0.f, Z = 0.f;
  const float* pb = part + (size_t)b * WPB * PART_F;
  for (int w = 0; w < WPB; ++w) {
    const float* p = pb + (size_t)w * PART_F;
    S += p[d * 16 + e];
    Z += p[256 + d] + p[256 + 16 + d];
  }
  ctx[d][e] = S / Z;
  __syncthreads();
  // B frag for attn WMMA (K zero-padded 16->32): halves j<8 -> K=j+8*(l>>4), j>=8 hold K>=16 -> 0
  for (int i = t; i < 512; i += 256) {
    int l = i >> 4, j = i & 15;
    ctxf[(size_t)b * 512 + i] = (_Float16)((j < 8) ? ctx[j + ((l & 16) ? 8 : 0)][l & 15] : 0.f);
  }
}

// ---------------- per-layer fused: q/softmax/attn/o/res/LN2/FF/res ----------------
#define NFRAG 21  // 0,1:wq(head) 2:ctx 3,4:wo(nt) 5..12:w1(nt) 13..20:w2(kstep*2+nt)
__global__ __launch_bounds__(128) void k_blk(float* __restrict__ hbuf,
    const float* __restrict__ ln1g, const float* __restrict__ ln1b,
    const float* __restrict__ wq, const float* __restrict__ wo, const float* __restrict__ bo,
    const float* __restrict__ ln2g, const float* __restrict__ ln2b,
    const float* __restrict__ w1, const float* __restrict__ b1,
    const float* __restrict__ w2, const float* __restrict__ b2,
    const _Float16* __restrict__ ctxf) {
  __shared__ v16h wfr[NFRAG][32];
  __shared__ float prm[320];
  __shared__ __align__(16) _Float16 ystage[NWAVE][1024];
  __shared__ float fstage[NWAVE][1024];
  __shared__ __align__(16) _Float16 gstage[NWAVE][2048];
  const int tid = threadIdx.x;
  const int b = blockIdx.y;
  for (int i = tid; i < NFRAG * 512; i += 128) {
    int f = i >> 9, e = i & 511, l2 = e >> 4, j = e & 15;
    int n = l2 & 15, kk = (j < 8 ? j : j + 8) + ((l2 & 16) ? 8 : 0);  // B-layout K index
    _Float16 hv;
    if (f == 2) hv = ctxf[(size_t)b * 512 + e];
    else {
      float vv;
      if (f < 2)       vv = wq[kk * 32 + f * 16 + n];
      else if (f < 5)  vv = wo[kk * 32 + (f - 3) * 16 + n];
      else if (f < 13) vv = w1[kk * 128 + (f - 5) * 16 + n];
      else { int idx = f - 13; vv = w2[((idx >> 1) * 32 + kk) * 32 + (idx & 1) * 16 + n]; }
      hv = (_Float16)vv;
    }
    ((_Float16*)&wfr[f][l2])[j] = hv;
  }
  for (int i = tid; i < 320; i += 128) {
    float vv;
    if (i < 32)       vv = ln1g[i];
    else if (i < 64)  vv = ln1b[i - 32];
    else if (i < 96)  vv = ln2g[i - 64];
    else if (i < 128) vv = ln2b[i - 96];
    else if (i < 160) vv = bo[i - 128];
    else if (i < 192) vv = b2[i - 160];
    else              vv = b1[i - 192];
    prm[i] = vv;
  }
  __syncthreads();
  const int w = tid >> 5, l = tid & 31, col = l & 15, rb = (l >> 4) << 3;
  _Float16* ys = ystage[w];
  float*    fs = fstage[w];
  _Float16* gs = gstage[w];
  const int wg = blockIdx.x * NWAVE + w;
  for (int t = wg; t < NTILE; t += WPB) {       // exactly 4 uniform iterations
    float* hp = hbuf + ((size_t)b * NSEQ + (size_t)t * 32 + l) * 32;
    float h0[32], y[32];
#pragma unroll
    for (int d = 0; d < 32; ++d) h0[d] = hp[d];
    layer_norm32(h0, y, prm, prm + 32);
    store_f16_row(ys + l * 32, y);
    __syncthreads();
    // q = y @ wq   (2 heads)
#pragma unroll
    for (int mt = 0; mt < 2; ++mt) {
      v16h a = load_a_frag(ys, mt * 16 + col, 32, 0, l);
#pragma unroll
      for (int hh = 0; hh < 2; ++hh) {
        v8f zc = {};
        v8f q = wmma_f16(a, wfr[hh][l], zc);
#pragma unroll
        for (int r = 0; r < 8; ++r) fs[(mt * 16 + rb + r) * 32 + hh * 16 + col] = q[r];
      }
    }
    __syncthreads();
    // softmax over feature dim per head, * DH^-0.5
    {
      const float* qq = fs + l * 32;
#pragma unroll
      for (int hh = 0; hh < 2; ++hh) {
        float mx = qq[hh * 16];
#pragma unroll
        for (int d = 1; d < 16; ++d) mx = fmaxf(mx, qq[hh * 16 + d]);
        float ex[16]; float s = 0.f;
#pragma unroll
        for (int d = 0; d < 16; ++d) { ex[d] = __expf(qq[hh * 16 + d] - mx); s += ex[d]; }
        float inv = 0.25f / s;
#pragma unroll
        for (int d = 0; d < 16; ++d) ys[l * 32 + hh * 16 + d] = (_Float16)(ex[d] * inv);
      }
    }
    __syncthreads();
    // attn = qs @ ctx (K zero-padded to 32)
    {
      v16h cf = wfr[2][l];
#pragma unroll
      for (int mt = 0; mt < 2; ++mt) {
#pragma unroll
        for (int hh = 0; hh < 2; ++hh) {
          v8h lo = *(const v8h*)(ys + (mt * 16 + col) * 32 + hh * 16 + ((l & 16) ? 8 : 0));
          v8h zz = {};
          v16h a = __builtin_shufflevector(lo, zz, 0,1,2,3,4,5,6,7,8,9,10,11,12,13,14,15);
          v8f zc = {};
          v8f at = wmma_f16(a, cf, zc);
#pragma unroll
          for (int r = 0; r < 8; ++r) gs[(mt * 16 + rb + r) * 32 + hh * 16 + col] = (_Float16)at[r];
        }
      }
    }
    __syncthreads();
    // o = attn @ wo
#pragma unroll
    for (int mt = 0; mt < 2; ++mt) {
      v16h a = load_a_frag(gs, mt * 16 + col, 32, 0, l);
#pragma unroll
      for (int nt = 0; nt < 2; ++nt) {
        v8f zc = {};
        v8f o = wmma_f16(a, wfr[3 + nt][l], zc);
#pragma unroll
        for (int r = 0; r < 8; ++r) fs[(mt * 16 + rb + r) * 32 + nt * 16 + col] = o[r];
      }
    }
    __syncthreads();
    // residual + LN2
    float h1[32], y2[32];
#pragma unroll
    for (int d = 0; d < 32; ++d) h1[d] = h0[d] + fs[l * 32 + d] + prm[128 + d];
    layer_norm32(h1, y2, prm + 64, prm + 96);
    store_f16_row(ys + l * 32, y2);
    __syncthreads();
    // FF: gelu(y2@w1 + b1) @ w2
#pragma unroll
    for (int mt = 0; mt < 2; ++mt) {
      v16h a2 = load_a_frag(ys, mt * 16 + col, 32, 0, l);
#pragma unroll
      for (int nt = 0; nt < 8; ++nt) {
        v8f zc = {};
        v8f f1 = wmma_f16(a2, wfr[5 + nt][l], zc);
        float bbias = prm[192 + nt * 16 + col];
#pragma unroll
        for (int r = 0; r < 8; ++r) {
          float xg = f1[r] + bbias;
          float gl = 0.5f * xg * (1.f + erff(xg * 0.70710678118654752f));  // exact gelu
          gs[(rb + r) * 128 + nt * 16 + col] = (_Float16)gl;
        }
      }
      __syncthreads();
      v8f f2a = {}, f2b = {};
#pragma unroll
      for (int ks = 0; ks < 4; ++ks) {   // K=128 as 4 chained WMMAs
        v16h ga = load_a_frag(gs, col, 128, ks * 32, l);
        f2a = wmma_f16(ga, wfr[13 + ks * 2][l],     f2a);
        f2b = wmma_f16(ga, wfr[13 + ks * 2 + 1][l], f2b);
      }
#pragma unroll
      for (int r = 0; r < 8; ++r) {
        fs[(mt * 16 + rb + r) * 32 + col]      = f2a[r];
        fs[(mt * 16 + rb + r) * 32 + 16 + col] = f2b[r];
      }
      __syncthreads();
    }
#pragma unroll
    for (int d = 0; d < 32; ++d) hp[d] = h1[d] + fs[l * 32 + d] + prm[160 + d];
  }
}

// ---------------- output: out[t] = h[t] . w_out + b_out ----------------
__global__ __launch_bounds__(256) void k_out(const float* __restrict__ hbuf,
                                             const float* __restrict__ wout,
                                             const float* __restrict__ bout,
                                             float* __restrict__ out) {
  size_t t = (size_t)blockIdx.x * 256 + threadIdx.x;
  if (t < (size_t)NBATCH * NSEQ) {
    const float* hp = hbuf + t * 32;
    float s = bout[0];
#pragma unroll
    for (int d = 0; d < 32; ++d) s += hp[d] * wout[d];
    out[t] = s;
  }
}

extern "C" void kernel_launch(void* const* d_in, const int* in_sizes, int n_in,
                              void* d_out, int out_size, void* d_ws, size_t ws_size,
                              hipStream_t stream) {
  (void)in_sizes; (void)n_in; (void)out_size; (void)ws_size;
  const float* x     = (const float*)d_in[0];
  const float* w_in  = (const float*)d_in[1];
  const float* b_in  = (const float*)d_in[2];
  const float* w_out = (const float*)d_in[3];
  const float* b_out = (const float*)d_in[4];
  const float* ln1_g = (const float*)d_in[5];
  const float* ln1_b = (const float*)d_in[6];
  const float* wq    = (const float*)d_in[7];
  const float* wk    = (const float*)d_in[8];
  const float* wv    = (const float*)d_in[9];
  const float* wo    = (const float*)d_in[10];
  const float* bo    = (const float*)d_in[11];
  const float* ln2_g = (const float*)d_in[12];
  const float* ln2_b = (const float*)d_in[13];
  const float* w1    = (const float*)d_in[14];
  const float* b1    = (const float*)d_in[15];
  const float* w2    = (const float*)d_in[16];
  const float* b2    = (const float*)d_in[17];
  float* out = (float*)d_out;

  char* ws = (char*)d_ws;
  float*    hbuf = (float*)ws;                                    // 32 MB
  float*    part = (float*)(ws + (size_t)33554432);               // 2.25 MB
  _Float16* ctxf = (_Float16*)(ws + (size_t)33554432 + 2359296);  // 4 KB

  k_embed<<<dim3(32768), dim3(256), 0, stream>>>(x, w_in, b_in, hbuf);
  for (int L = 0; L < 4; ++L) {
    k_ctx<<<dim3(BLKX, NBATCH), dim3(128), 0, stream>>>(
        hbuf, ln1_g + L * 32, ln1_b + L * 32, wk + L * 512, wv + L * 512, part);
    k_fin<<<dim3(NBATCH), dim3(256), 0, stream>>>(part, ctxf);
    k_blk<<<dim3(BLKX, NBATCH), dim3(128), 0, stream>>>(
        hbuf, ln1_g + L * 32, ln1_b + L * 32, wq + L * 1024, wo + L * 1024, bo + L * 32,
        ln2_g + L * 32, ln2_b + L * 32, w1 + L * 4096, b1 + L * 128, w2 + L * 4096, b2 + L * 32,
        ctxf);
  }
  k_out<<<dim3(1024), dim3(256), 0, stream>>>(hbuf, w_out, b_out, out);
}
